// RNN_74672301408412
// MI455X (gfx1250) — compile-verified, matrix operand order checked by reading, not run on hardware
//
#include <hip/hip_runtime.h>
#include <math.h>
#include <stdint.h>

// ---------------- problem constants ----------------
#define BB 64
#define TT 512
#define VV 8192
#define EE 1024
#define HH 2048

typedef __bf16 bfx;
typedef bfx      v16bf __attribute__((ext_vector_type(16)));
typedef float    v8f   __attribute__((ext_vector_type(8)));
typedef uint32_t u32x4 __attribute__((ext_vector_type(4)));
typedef int      i32x4 __attribute__((ext_vector_type(4)));
typedef int      i32x8 __attribute__((ext_vector_type(8)));

#if defined(__has_builtin)
#  if __has_builtin(__builtin_amdgcn_tensor_load_to_lds)
#    define HAS_TDM 1
#  endif
#endif
#ifndef HAS_TDM
#  define HAS_TDM 0
#endif

// ---------------- WMMA core ----------------
__device__ __forceinline__ v8f wmma_bf16(v16bf a, v16bf b, v8f c) {
    return __builtin_amdgcn_wmma_f32_16x16x32_bf16(
        false, a, false, b, (short)0, c, false, false);
}

// A/B fragment from a row-major bf16 matrix [rows][ld] (ld in elements):
// lane l -> row r0+(l&15); K chunks at k0+(l>>4)*8 and +16 (ISA 7.12.2 layout)
template <typename P>
__device__ __forceinline__ v16bf frag_ld_bf16(P base, int ld, int r0, int k0, int lane) {
    auto p = base + (size_t)(r0 + (lane & 15)) * ld + (k0 + ((lane >> 4) << 3));
    union { v16bf v; uint4 q[2]; } u;
    u.q[0] = *(const uint4*)(p);
    u.q[1] = *(const uint4*)(p + 16);
    return u.v;
}

// A fragment gathered from an fp32 row pointer (embedding gather), cvt to bf16
__device__ __forceinline__ v16bf frag_ld_f32cvt(const float* __restrict__ rowp,
                                                int k0, int lane) {
    int kb = k0 + ((lane >> 4) << 3);
    v16bf v;
#pragma unroll
    for (int i = 0; i < 8; ++i) v[i] = (bfx)rowp[kb + i];
#pragma unroll
    for (int i = 0; i < 8; ++i) v[i + 8] = (bfx)rowp[kb + 16 + i];
    return v;
}

// ---------------- TDM: 2D bf16 tile -> padded LDS ----------------
// Loads tile_d1 rows x tile_d0 elements from a row-major bf16 tensor
// (row stride = row_stride_elems) into LDS at lds_byte_addr.
// Padding: 4 DWORDs after every 16 DWORDs -> LDS row stride 80B (40 elems).
#define LDSP 40   // padded LDS row stride in bf16 elements (80 bytes)

__device__ __forceinline__ void tdm_load_tile_bf16(
    uint32_t lds_byte_addr, const bfx* gsrc,
    uint32_t tensor_d0, uint32_t tensor_d1, uint64_t row_stride_elems,
    uint32_t tile_d0, uint32_t tile_d1)
{
#if HAS_TDM
    uint64_t ga = (uint64_t)(uintptr_t)gsrc;
    u32x4 g0 = {};
    g0[0] = 1u;                                            // count=1 (valid D#)
    g0[1] = lds_byte_addr;                                 // lds_addr
    g0[2] = (uint32_t)ga;                                  // global_addr[31:0]
    g0[3] = ((uint32_t)(ga >> 32) & 0x01FFFFFFu)           // global_addr[56:32]
          | (2u << 30);                                    // type=2 ("image")
    i32x8 g1 = {};
    g1[0] = (int)((1u << 16)      // data_size = 1 -> 2 bytes (bf16)
                | (1u << 20)      // pad_enable
                | (3u << 22)      // pad_interval code 3 -> every 16 DWORDs (64B row)
                | (3u << 25));    // pad_amount   code 3 -> 4 DWORDs (16B)
    g1[1] = (int)((tensor_d0 & 0xFFFFu) << 16);            // tensor_dim0[15:0]
    g1[2] = (int)(((tensor_d0 >> 16) & 0xFFFFu) | ((tensor_d1 & 0xFFFFu) << 16));
    g1[3] = (int)(((tensor_d1 >> 16) & 0xFFFFu) | ((tile_d0 & 0xFFFFu) << 16));
    g1[4] = (int)(tile_d1 & 0xFFFFu);                      // tile_dim1; tile_dim2=0
    g1[5] = (int)(uint32_t)row_stride_elems;               // tensor_dim0_stride lo
    g1[6] = (int)((uint32_t)(row_stride_elems >> 32) & 0xFFFFu);
    g1[7] = 0;
    i32x4 g2 = {};
    i32x4 g3 = {};
#  if __clang_major__ >= 23
    i32x8 g4 = {};
    __builtin_amdgcn_tensor_load_to_lds(g0, g1, g2, g3, g4, 0);
#  else
    __builtin_amdgcn_tensor_load_to_lds(g0, g1, g2, g3, 0);
#  endif
#endif
}

// ---------------- kernel 0: transpose + fp32->bf16 convert ----------------
__global__ __launch_bounds__(256)
void transpose_cvt(const float* __restrict__ src, bfx* __restrict__ dst,
                   int R, int C) {
    __shared__ float tile[32][33];
    int c0 = blockIdx.x * 32, r0 = blockIdx.y * 32;
#pragma unroll
    for (int i = threadIdx.y; i < 32; i += 8) {
        int r = r0 + i, c = c0 + threadIdx.x;
        tile[i][threadIdx.x] = (r < R && c < C) ? src[(size_t)r * C + c] : 0.0f;
    }
    __syncthreads();
#pragma unroll
    for (int i = threadIdx.y; i < 32; i += 8) {
        int rr = c0 + i, cc = r0 + threadIdx.x;
        if (rr < C && cc < R) dst[(size_t)rr * R + cc] = (bfx)tile[threadIdx.x][i];
    }
}

// ---------------- init kernels ----------------
__global__ void h0_init(const float* __restrict__ start, bfx* __restrict__ h0) {
    int i = blockIdx.x * blockDim.x + threadIdx.x;
    if (i < BB * HH) h0[i] = (bfx)start[i % HH];
}
__global__ void zero_accum(float* a) { if (threadIdx.x < 2) a[threadIdx.x] = 0.0f; }

// ---------------- kernel 1: xproj = gather(W_E, idx) @ Wx + b_xh -----------
__global__ __launch_bounds__(256)
void xproj_gemm(const int* __restrict__ idx, const float* __restrict__ W_E,
                const bfx* __restrict__ WxT /*H x E*/,
                const float* __restrict__ b_xh, bfx* __restrict__ xproj) {
    const int lane = threadIdx.x & 31;
    const int w    = threadIdx.x >> 5;
    const int m0 = blockIdx.x * 32 + (w & 1) * 16;
    const int n0 = blockIdx.y * 256 + (w >> 1) * 64;

    const int m = m0 + (lane & 15);
    const int t = m >> 6, b = m & 63;
    const float* rowp = W_E + (size_t)idx[b * TT + t] * EE;

    v8f acc[4];
#pragma unroll
    for (int j = 0; j < 4; ++j) acc[j] = (v8f){};

    for (int k = 0; k < EE; k += 32) {
        __builtin_prefetch(rowp + k + 32, 0, 0);
        v16bf a = frag_ld_f32cvt(rowp, k, lane);
#pragma unroll
        for (int j = 0; j < 4; ++j) {
            v16bf bf = frag_ld_bf16(WxT, EE, n0 + j * 16, k, lane);
            acc[j] = wmma_bf16(a, bf, acc[j]);
        }
    }
    const int rbase = m0 + ((lane >> 4) << 3);
#pragma unroll
    for (int j = 0; j < 4; ++j) {
        const int n = n0 + j * 16 + (lane & 15);
        const float bias = b_xh[n];
#pragma unroll
        for (int r = 0; r < 8; ++r)
            xproj[(size_t)(rbase + r) * HH + n] = (bfx)(acc[j][r] + bias);
    }
}

// ---------------- kernel 2: one recurrent step -----------------------------
__global__ __launch_bounds__(256)
void rnn_step(const bfx* __restrict__ hprev, const bfx* __restrict__ xproj_t,
              const bfx* __restrict__ WhT, bfx* __restrict__ ht) {
    const int lane = threadIdx.x & 31;
    const int W = blockIdx.x * 8 + (threadIdx.x >> 5);
    const int m0 = (W & 3) * 16;
    const int n0 = (W >> 2) * 32;

    v8f acc0 = (v8f){}, acc1 = (v8f){};
    for (int k = 0; k < HH; k += 32) {
        v16bf a  = frag_ld_bf16(hprev, HH, m0, k, lane);
        v16bf b0 = frag_ld_bf16(WhT, HH, n0, k, lane);
        v16bf b1 = frag_ld_bf16(WhT, HH, n0 + 16, k, lane);
        acc0 = wmma_bf16(a, b0, acc0);
        acc1 = wmma_bf16(a, b1, acc1);
    }
    const int rbase = m0 + ((lane >> 4) << 3);
#pragma unroll
    for (int j = 0; j < 2; ++j) {
        const v8f acc = j ? acc1 : acc0;
        const int n = n0 + j * 16 + (lane & 15);
#pragma unroll
        for (int r = 0; r < 8; ++r) {
            const int row = rbase + r;
            float v = acc[r] + (float)xproj_t[(size_t)row * HH + n];
            ht[(size_t)row * HH + n] = (bfx)tanhf(v);
        }
    }
}

// ---------------- kernel 3: head GEMM via TDM double-buffered LDS ----------
// M = T*B (m = t*B+b), N = V, K = H. Block: 8 waves -> 32M x 256N tile.
// Per K-chunk (32): TDM stages B tile (256x32) and A tile (32x32) into padded
// LDS; waves run 4 WMMAs each off LDS while TDM prefetches the next chunk.
__global__ __launch_bounds__(256)
void head_gemm(const bfx* __restrict__ hidden, const bfx* __restrict__ WheadT,
               const float* __restrict__ b_head, float* __restrict__ logits) {
    __shared__ __align__(128) bfx ldsB[2][256 * LDSP];  // 2 x 20 KiB
    __shared__ __align__(128) bfx ldsA[2][32 * LDSP];   // 2 x 2.5 KiB

    const int lane = threadIdx.x & 31;
    const int w    = threadIdx.x >> 5;
    const int mBase = blockIdx.x * 32;
    const int nBase = blockIdx.y * 256;
    const int m0w = (w & 1) * 16;        // wave row offset within A tile
    const int n0w = (w >> 1) * 64;       // wave col offset within B tile

    v8f acc[4];
#pragma unroll
    for (int j = 0; j < 4; ++j) acc[j] = (v8f){};

    const int nIter = HH / 32;

#if HAS_TDM
    // prologue: stage chunk 0 into buffer 0 (wave0 -> B tile, wave1 -> A tile)
    if (w == 0)
        tdm_load_tile_bf16((uint32_t)(uintptr_t)&ldsB[0][0],
                           WheadT + (size_t)nBase * HH, HH, VV, HH, 32, 256);
    else if (w == 1)
        tdm_load_tile_bf16((uint32_t)(uintptr_t)&ldsA[0][0],
                           hidden + (size_t)mBase * HH, HH, TT * BB, HH, 32, 32);

    for (int it = 0; it < nIter; ++it) {
        const int buf = it & 1;
        if (it + 1 < nIter) {               // prefetch next chunk into other buf
            const int k = (it + 1) * 32;
            if (w == 0)
                tdm_load_tile_bf16((uint32_t)(uintptr_t)&ldsB[buf ^ 1][0],
                                   WheadT + (size_t)nBase * HH + k, HH, VV, HH, 32, 256);
            else if (w == 1)
                tdm_load_tile_bf16((uint32_t)(uintptr_t)&ldsA[buf ^ 1][0],
                                   hidden + (size_t)mBase * HH + k, HH, TT * BB, HH, 32, 32);
        }
        if (w < 2) {                        // current buffer complete (TDM in-order)
            if (it + 1 < nIter)
                __builtin_amdgcn_s_wait_tensorcnt(1);   // one prefetch outstanding
            else
                __builtin_amdgcn_s_wait_tensorcnt(0);   // drain on last chunk
        }
        __syncthreads();                    // publish staged tiles to all waves

        v16bf a = frag_ld_bf16((const bfx*)&ldsA[buf][0], LDSP, m0w, 0, lane);
#pragma unroll
        for (int j = 0; j < 4; ++j) {
            v16bf bf = frag_ld_bf16((const bfx*)&ldsB[buf][0], LDSP, n0w + j * 16, 0, lane);
            acc[j] = wmma_bf16(a, bf, acc[j]);
        }
        __syncthreads();                    // reads done before buffer is reissued
    }
#else
    // fallback: cooperative synchronous staging, single buffer
    for (int it = 0; it < nIter; ++it) {
        const int k = it * 32;
        __syncthreads();
        {   // 256 threads: one B row each; threads 0..31 also one A row each
            const int r = threadIdx.x;
            const uint4* src = (const uint4*)(WheadT + (size_t)(nBase + r) * HH + k);
            uint4* dst = (uint4*)&ldsB[0][r * LDSP];
            dst[0] = src[0]; dst[1] = src[1];
            if (r < 32) {
                const uint4* sa = (const uint4*)(hidden + (size_t)(mBase + r) * HH + k);
                uint4* da = (uint4*)&ldsA[0][r * LDSP];
                da[0] = sa[0]; da[1] = sa[1];
            }
        }
        __syncthreads();
        v16bf a = frag_ld_bf16((const bfx*)&ldsA[0][0], LDSP, m0w, 0, lane);
#pragma unroll
        for (int j = 0; j < 4; ++j) {
            v16bf bf = frag_ld_bf16((const bfx*)&ldsB[0][0], LDSP, n0w + j * 16, 0, lane);
            acc[j] = wmma_bf16(a, bf, acc[j]);
        }
    }
#endif

    // epilogue: + bias, remap m = t*B+b -> logits[b][t][v], fp32 store
    const int m0 = mBase + m0w;
    const int rbase = m0 + ((lane >> 4) << 3);
#pragma unroll
    for (int j = 0; j < 4; ++j) {
        const int n = nBase + n0w + j * 16 + (lane & 15);
        const float bias = b_head[n];
#pragma unroll
        for (int r = 0; r < 8; ++r) {
            const int m = rbase + r;
            const int t = m >> 6, b = m & 63;
            logits[(size_t)(b * TT + t) * VV + n] = acc[j][r] + bias;
        }
    }
}

// ---------------- kernel 4: per-row log-softmax NLL ------------------------
__global__ __launch_bounds__(256)
void loss_rows(const float* __restrict__ logits, const int* __restrict__ targets,
               float* __restrict__ accum) {
    const int row = blockIdx.x;
    const float* lp = logits + (size_t)row * VV;
    __shared__ float red[256];

    float mx = -3.0e38f;
    for (int i = threadIdx.x; i < VV; i += 256) mx = fmaxf(mx, lp[i]);
    red[threadIdx.x] = mx; __syncthreads();
    for (int s = 128; s > 0; s >>= 1) {
        if (threadIdx.x < s) red[threadIdx.x] = fmaxf(red[threadIdx.x], red[threadIdx.x + s]);
        __syncthreads();
    }
    mx = red[0]; __syncthreads();

    float sum = 0.0f;
    for (int i = threadIdx.x; i < VV; i += 256) sum += __expf(lp[i] - mx);
    red[threadIdx.x] = sum; __syncthreads();
    for (int s = 128; s > 0; s >>= 1) {
        if (threadIdx.x < s) red[threadIdx.x] += red[threadIdx.x + s];
        __syncthreads();
    }
    if (threadIdx.x == 0) {
        const float lse = mx + __logf(red[0]);
        const int tgt = targets[row];
        const bool maskv = (tgt != -1);
        const float nll = maskv ? (lse - lp[maskv ? tgt : 0]) : 0.0f;
        atomicAdd(&accum[0], nll);
        atomicAdd(&accum[1], maskv ? 1.0f : 0.0f);
    }
}

__global__ void finalize_loss(const float* __restrict__ accum, float* __restrict__ out) {
    out[0] = accum[0] / fmaxf(accum[1], 1.0f);
}

// ---------------- host-side orchestration ----------------
extern "C" void kernel_launch(void* const* d_in, const int* in_sizes, int n_in,
                              void* d_out, int out_size, void* d_ws, size_t ws_size,
                              hipStream_t stream) {
    const int*   idx     = (const int*)  d_in[0];
    const int*   targets = (const int*)  d_in[1];
    const float* W_E     = (const float*)d_in[2];
    const float* start   = (const float*)d_in[3];
    const float* W_xh    = (const float*)d_in[4];
    const float* b_xh    = (const float*)d_in[5];
    const float* W_head  = (const float*)d_in[6];
    const float* b_head  = (const float*)d_in[7];
    float* out = (float*)d_out;

    char* cur = (char*)d_ws;
    auto carve = [&](size_t bytes) -> void* {
        void* p = (void*)cur;
        cur += (bytes + 255) & ~(size_t)255;
        return p;
    };
    bfx* WxT    = (bfx*)carve((size_t)HH * EE * sizeof(bfx));
    bfx* WhT    = (bfx*)carve((size_t)HH * HH * sizeof(bfx));
    bfx* WheadT = (bfx*)carve((size_t)VV * HH * sizeof(bfx));
    bfx* xproj  = (bfx*)carve((size_t)TT * BB * HH * sizeof(bfx));
    bfx* hidden = (bfx*)carve((size_t)TT * BB * HH * sizeof(bfx));
    bfx* h0     = (bfx*)carve((size_t)BB * HH * sizeof(bfx));
    float* accum = (float*)carve(2 * sizeof(float));

    transpose_cvt<<<dim3(HH / 32, EE / 32), dim3(32, 8), 0, stream>>>(W_xh, WxT, EE, HH);
    transpose_cvt<<<dim3(HH / 32, HH / 32), dim3(32, 8), 0, stream>>>(W_xh + (size_t)EE * HH, WhT, HH, HH);
    transpose_cvt<<<dim3(VV / 32, HH / 32), dim3(32, 8), 0, stream>>>(W_head, WheadT, HH, VV);
    h0_init<<<(BB * HH + 255) / 256, 256, 0, stream>>>(start, h0);
    zero_accum<<<1, 32, 0, stream>>>(accum);

    xproj_gemm<<<dim3((TT * BB) / 32, HH / 256), 256, 0, stream>>>(idx, W_E, WxT, b_xh, xproj);

    const bfx* hprev = h0;
    for (int t = 0; t < TT; ++t) {
        bfx* ht = hidden + (size_t)t * BB * HH;
        const bfx* xp = xproj + (size_t)t * BB * HH;
        rnn_step<<<32, 256, 0, stream>>>(hprev, xp, WhT, ht);
        hprev = ht;
    }

    head_gemm<<<dim3((TT * BB) / 32, VV / 256), 256, 0, stream>>>(hidden, WheadT, b_head, out);

    loss_rows<<<BB * TT, 256, 0, stream>>>(out, targets, accum);
    finalize_loss<<<1, 1, 0, stream>>>(accum, out + (size_t)BB * TT * VV);
}
